// MambaDiffusion_8899172237907
// MI455X (gfx1250) — compile-verified, matrix operand order checked by reading, not run on hardware
//
#include <hip/hip_runtime.h>
#include <hip/hip_bf16.h>
#include <stdint.h>

// ---------------------------------------------------------------------------
// Model constants (match reference)
// ---------------------------------------------------------------------------
#define BB   16
#define IMG  64
#define CC   3
#define PP   4
#define DD   256
#define NLAY 4
#define DI   512
#define DS   16
#define DR   32
#define LL   256          // (IMG/P)^2
#define NTOK (BB * LL)    // 4096

typedef __bf16 bf16;
typedef __attribute__((ext_vector_type(8)))  __bf16 v8bf;
typedef __attribute__((ext_vector_type(16))) __bf16 v16bf;
typedef __attribute__((ext_vector_type(8)))  float  v8f;

// ---------------------------------------------------------------------------
// Helpers
// ---------------------------------------------------------------------------
__device__ __forceinline__ bf16 f2bf(float f) {
    unsigned u = __builtin_bit_cast(unsigned, f);
    unsigned r = u + 0x7FFFu + ((u >> 16) & 1u);   // round-to-nearest-even
    unsigned short hs = (unsigned short)(r >> 16);
    return __builtin_bit_cast(bf16, hs);
}
__device__ __forceinline__ float siluf(float x) {
    return x / (1.0f + __expf(-x));
}
__device__ __forceinline__ float softplusf(float x) {
    return (x > 20.0f) ? x : __logf(1.0f + __expf(x));
}

// Lane fragment slice: 8 contiguous bf16 at p (16B) and 8 at p+16 (16B).
__device__ __forceinline__ v16bf load_frag(const bf16* __restrict__ p) {
    v8bf lo = *(const v8bf*)p;
    v8bf hi = *(const v8bf*)(p + 16);
    return __builtin_shufflevector(lo, hi, 0, 1, 2, 3, 4, 5, 6, 7,
                                           8, 9, 10, 11, 12, 13, 14, 15);
}

// ---------------------------------------------------------------------------
// fp32 -> bf16 bulk conversion (n multiple of 4)
// ---------------------------------------------------------------------------
__global__ __launch_bounds__(256)
void cvt_bf16_kernel(const float* __restrict__ in, bf16* __restrict__ out, int n)
{
    int idx = (blockIdx.x * 256 + threadIdx.x) * 4;
    if (idx + 3 < n) {
#pragma unroll
        for (int j = 0; j < 4; ++j) out[idx + j] = f2bf(in[idx + j]);
    }
}

// ---------------------------------------------------------------------------
// WMMA GEMM:  C[M,N] = act( A[M,K] * W[N,K]^T + bias ),  A/W bf16, C fp32
// Optional secondary bf16 output Cbf. One wave computes a 16(M) x NT*16(N)
// tile. Ping-pong double-buffered fragments (two disjoint register sets) ->
// no inner-loop register copies, one k-step of load-latency hiding.
// Grid: x = ceil(N/(NT*16)), y = ceil(M/64); block = 128 (4 waves).
// act: 0 = none, 1 = softplus
// ---------------------------------------------------------------------------
template <int NT>
__global__ __launch_bounds__(128)
void gemm_wmma_kernel(const bf16* __restrict__ A, int lda,
                      const bf16* __restrict__ W, int ldw,
                      float* __restrict__ C, bf16* __restrict__ Cbf, int ldc,
                      int M, int N, int K,
                      const float* __restrict__ bias, int act)
{
    const int lane = threadIdx.x & 31;
    const int wave = threadIdx.x >> 5;
    const int m0   = (blockIdx.y * 4 + wave) * 16;
    if (m0 >= M) return;
    const int n0base = blockIdx.x * (NT * 16);

    const int half = lane >> 4;     // 0: K 0-7/16-23 ; 1: K 8-15/24-31
    const int r    = lane & 15;     // row (A) / col (B) within tile

    const bf16* arow = A + (size_t)(m0 + r) * lda + half * 8;
    const bf16* wrow[NT];
#pragma unroll
    for (int nt = 0; nt < NT; ++nt)
        wrow[nt] = W + (size_t)(n0base + nt * 16 + r) * ldw + half * 8;

    v8f acc[NT];
#pragma unroll
    for (int nt = 0; nt < NT; ++nt) acc[nt] = v8f{};

    // buffer 0: fragments for k = 0
    v16bf a0 = load_frag(arow);
    v16bf b0[NT];
#pragma unroll
    for (int nt = 0; nt < NT; ++nt) b0[nt] = load_frag(wrow[nt]);

    v16bf a1;
    v16bf b1[NT];

    int k = 0;
    // steady state: two k-steps per iteration, ping-pong buffers, no copies
    while (k + 64 <= K) {
        // prefetch k+32 into buf1, compute buf0 (k)
        a1 = load_frag(arow + k + 32);
#pragma unroll
        for (int nt = 0; nt < NT; ++nt) b1[nt] = load_frag(wrow[nt] + k + 32);
#pragma unroll
        for (int nt = 0; nt < NT; ++nt)
            acc[nt] = __builtin_amdgcn_wmma_f32_16x16x32_bf16(
                false, a0, false, b0[nt], (short)0, acc[nt], false, false);

        // prefetch k+64 into buf0 (if it exists), compute buf1 (k+32)
        if (k + 64 < K) {
            a0 = load_frag(arow + k + 64);
#pragma unroll
            for (int nt = 0; nt < NT; ++nt)
                b0[nt] = load_frag(wrow[nt] + k + 64);
        }
#pragma unroll
        for (int nt = 0; nt < NT; ++nt)
            acc[nt] = __builtin_amdgcn_wmma_f32_16x16x32_bf16(
                false, a1, false, b1[nt], (short)0, acc[nt], false, false);
        k += 64;
    }
    // remainder single k-step (K/32 odd, e.g. K == 32)
    if (k < K) {
#pragma unroll
        for (int nt = 0; nt < NT; ++nt)
            acc[nt] = __builtin_amdgcn_wmma_f32_16x16x32_bf16(
                false, a0, false, b0[nt], (short)0, acc[nt], false, false);
    }

    // store: lane holds column n = n0 + r, rows m = half*8 + v
#pragma unroll
    for (int nt = 0; nt < NT; ++nt) {
        int n = n0base + nt * 16 + r;
        float bv = bias ? bias[n] : 0.0f;
#pragma unroll
        for (int v = 0; v < 8; ++v) {
            int m = m0 + half * 8 + v;
            float val = acc[nt][v] + bv;
            if (act == 1) val = softplusf(val);
            C[(size_t)m * ldc + n] = val;
            if (Cbf) Cbf[(size_t)m * ldc + n] = f2bf(val);
        }
    }
}

static void launch_gemm(const bf16* A, int lda, const bf16* W, int ldw,
                        float* C, bf16* Cbf, int ldc, int M, int N, int K,
                        const float* bias, int act, hipStream_t s)
{
    dim3 block(128);
    if (N % 64 == 0) {
        dim3 grid(N / 64, (M + 63) / 64);
        gemm_wmma_kernel<4><<<grid, block, 0, s>>>(A, lda, W, ldw, C, Cbf, ldc,
                                                   M, N, K, bias, act);
    } else {            // only the head: N = 48 -> 3 subtiles, one x-block
        dim3 grid(1, (M + 63) / 64);
        gemm_wmma_kernel<3><<<grid, block, 0, s>>>(A, lda, W, ldw, C, Cbf, ldc,
                                                   M, N, K, bias, act);
    }
}

// ---------------------------------------------------------------------------
// Patch embedding: h[tok, d] = patch_b[d] + sum_j patch[j] * patch_w[d, j]
// ---------------------------------------------------------------------------
__global__ __launch_bounds__(256)
void patch_embed_kernel(const float* __restrict__ x,
                        const float* __restrict__ pw,
                        const float* __restrict__ pb,
                        float* __restrict__ h)
{
    __shared__ float patch[48];
    int tok = blockIdx.x;
    int d   = threadIdx.x;
    int b   = tok >> 8;
    int p   = tok & 255;
    int hp  = p >> 4, wp = p & 15;
    if (d < 48) {
        int c  = d >> 4;
        int py = (d >> 2) & 3;
        int px = d & 3;
        patch[d] = x[(((size_t)(b * CC + c) * IMG) + hp * PP + py) * IMG + wp * PP + px];
    }
    __syncthreads();
    float acc = pb[d];
    const float* wrow = pw + (size_t)d * 48;
#pragma unroll 8
    for (int j = 0; j < 48; ++j) acc += patch[j] * wrow[j];
    h[(size_t)tok * DD + d] = acc;
}

// ---------------------------------------------------------------------------
// Time embedding MLP: sinusoidal -> D -> silu -> D. One block per batch elem.
// ---------------------------------------------------------------------------
__global__ __launch_bounds__(256)
void time_embed_kernel(const int* __restrict__ t,
                       const float* __restrict__ tw1, const float* __restrict__ tb1,
                       const float* __restrict__ tw2, const float* __restrict__ tb2,
                       float* __restrict__ te)
{
    __shared__ float vec[DD];
    __shared__ float hid[DD];
    int b = blockIdx.x, d = threadIdx.x;
    float tf = (float)t[b];
    const float scale = -logf(10000.0f) / 127.0f;   // half-1 = 127
    if (d < 128) vec[d] = sinf(tf * expf((float)d * scale));
    else         vec[d] = cosf(tf * expf((float)(d - 128) * scale));
    __syncthreads();
    float a = tb1[d];
    const float* w1 = tw1 + (size_t)d * DD;
    for (int k = 0; k < DD; ++k) a += vec[k] * w1[k];
    hid[d] = siluf(a);
    __syncthreads();
    float o = tb2[d];
    const float* w2 = tw2 + (size_t)d * DD;
    for (int k = 0; k < DD; ++k) o += hid[k] * w2[k];
    te[(size_t)b * DD + d] = o;
}

// ---------------------------------------------------------------------------
// out_bf[tok,:] = bf16( LayerNorm( h[tok,:] + te[b,:] ) * g + beta )
// One block (256 thr = D) per token.  te may be null.
// ---------------------------------------------------------------------------
__global__ __launch_bounds__(256)
void add_ln_kernel(const float* __restrict__ h, const float* __restrict__ te,
                   const float* __restrict__ g, const float* __restrict__ beta,
                   bf16* __restrict__ out_bf)
{
    __shared__ float red[DD];
    int tok = blockIdx.x, d = threadIdx.x;
    float v = h[(size_t)tok * DD + d];
    if (te) v += te[(size_t)(tok >> 8) * DD + d];
    red[d] = v;
    __syncthreads();
    for (int s = 128; s > 0; s >>= 1) {
        if (d < s) red[d] += red[d + s];
        __syncthreads();
    }
    float mu = red[0] * (1.0f / DD);
    __syncthreads();
    float c = v - mu;
    red[d] = c * c;
    __syncthreads();
    for (int s = 128; s > 0; s >>= 1) {
        if (d < s) red[d] += red[d + s];
        __syncthreads();
    }
    float rstd = rsqrtf(red[0] * (1.0f / DD) + 1e-5f);
    out_bf[(size_t)tok * DD + d] = f2bf(c * rstd * g[d] + beta[d]);
}

// ---------------------------------------------------------------------------
// Depthwise causal conv (kernel 4, left pad 3) + bias + SiLU.
// xz layout [NTOK, 2*DI]; writes u fp32 (scan) and bf16 (GEMM input).
// ---------------------------------------------------------------------------
__global__ __launch_bounds__(256)
void conv_silu_kernel(const float* __restrict__ xz,
                      const float* __restrict__ cw, const float* __restrict__ cb,
                      float* __restrict__ u, bf16* __restrict__ ubf)
{
    int c   = blockIdx.x * 256 + threadIdx.x;
    int tok = blockIdx.y;
    int l   = tok & 255;
    float acc = cb[c];
    const float* w = cw + (size_t)c * 4;
#pragma unroll
    for (int j = 0; j < 4; ++j) {
        int li = l - 3 + j;
        if (li >= 0) acc += w[j] * xz[(size_t)(tok - 3 + j) * (2 * DI) + c];
    }
    float val = siluf(acc);
    u[(size_t)tok * DI + c]   = val;
    ubf[(size_t)tok * DI + c] = f2bf(val);
}

// ---------------------------------------------------------------------------
// Selective-scan. One thread per (batch, channel): 16 states in registers.
// proj[tok, 32:48]=Bc, [48:64]=Cc. backward: reverse time. accumulate: y +=.
// ---------------------------------------------------------------------------
__global__ __launch_bounds__(256)
void ssm_scan_kernel(const float* __restrict__ u, const float* __restrict__ dt,
                     const float* __restrict__ proj,
                     const float* __restrict__ Alog, const float* __restrict__ Dv,
                     float* __restrict__ y, int backward, int accumulate)
{
    int di = blockIdx.x * 256 + threadIdx.x;
    int b  = blockIdx.y;
    float A[DS];
#pragma unroll
    for (int s = 0; s < DS; ++s) A[s] = -__expf(Alog[(size_t)di * DS + s]);
    float hst[DS];
#pragma unroll
    for (int s = 0; s < DS; ++s) hst[s] = 0.0f;
    float dval = Dv[di];

    for (int step = 0; step < LL; ++step) {
        int l = backward ? (LL - 1 - step) : step;
        size_t tok = (size_t)b * LL + l;
        float ut  = u[tok * DI + di];
        float dtt = dt[tok * DI + di];
        float du  = dtt * ut;
        const float* Bc = proj + tok * 64 + 32;
        const float* Cv = proj + tok * 64 + 48;
        float acc = 0.0f;
#pragma unroll
        for (int s = 0; s < DS; ++s) {
            hst[s] = __expf(dtt * A[s]) * hst[s] + du * Bc[s];
            acc += hst[s] * Cv[s];
        }
        float val = acc + ut * dval;
        if (accumulate) y[tok * DI + di] += val;
        else            y[tok * DI + di]  = val;
    }
}

// ---------------------------------------------------------------------------
// ybf = bf16( y * silu(z) ),  z = xz[:, DI:2*DI]
// ---------------------------------------------------------------------------
__global__ __launch_bounds__(256)
void gate_kernel(const float* __restrict__ y, const float* __restrict__ xz,
                 bf16* __restrict__ ybf)
{
    int idx = blockIdx.x * 256 + threadIdx.x;     // < NTOK*DI
    int tok = idx >> 9, di = idx & 511;
    float z = xz[(size_t)tok * (2 * DI) + DI + di];
    ybf[idx] = f2bf(y[idx] * siluf(z));
}

__global__ __launch_bounds__(256)
void add_kernel(float* __restrict__ a, const float* __restrict__ b, int n)
{
    int idx = blockIdx.x * 256 + threadIdx.x;
    if (idx < n) a[idx] += b[idx];
}

// ---------------------------------------------------------------------------
// Unpatchify: o[tok, c*16+py*4+px] -> out[b, c, hp*4+py, wp*4+px]
// ---------------------------------------------------------------------------
__global__ __launch_bounds__(256)
void unpatch_kernel(const float* __restrict__ o, float* __restrict__ out)
{
    int idx = blockIdx.x * 256 + threadIdx.x;
    if (idx >= BB * CC * IMG * IMG) return;
    int col = idx & 63;
    int row = (idx >> 6) & 63;
    int c   = (idx >> 12) % CC;
    int b   = idx / (CC * IMG * IMG);
    int hp = row >> 2, py = row & 3;
    int wp = col >> 2, px = col & 3;
    out[idx] = o[(size_t)(b * LL + hp * 16 + wp) * 48 + c * 16 + py * 4 + px];
}

// ---------------------------------------------------------------------------
// Orchestration
// ---------------------------------------------------------------------------
static void cvt(const float* in, bf16* out, int n, hipStream_t s)
{
    cvt_bf16_kernel<<<(n / 4 + 255) / 256, 256, 0, s>>>(in, out, n);
}

extern "C" void kernel_launch(void* const* d_in, const int* in_sizes, int n_in,
                              void* d_out, int out_size, void* d_ws, size_t ws_size,
                              hipStream_t stream)
{
    const float* x        = (const float*)d_in[0];
    const int*   t        = (const int*)  d_in[1];
    const float* patch_w  = (const float*)d_in[2];
    const float* patch_b  = (const float*)d_in[3];
    const float* tw1      = (const float*)d_in[4];
    const float* tb1      = (const float*)d_in[5];
    const float* tw2      = (const float*)d_in[6];
    const float* tb2      = (const float*)d_in[7];
    const float* norm_g   = (const float*)d_in[8];
    const float* norm_b   = (const float*)d_in[9];
    const float* inproj_w = (const float*)d_in[10];
    const float* conv_w   = (const float*)d_in[11];
    const float* conv_b   = (const float*)d_in[12];
    const float* Alog_f   = (const float*)d_in[13];
    const float* D_f      = (const float*)d_in[14];
    const float* xproj_f  = (const float*)d_in[15];
    const float* dtw_f    = (const float*)d_in[16];
    const float* dtb_f    = (const float*)d_in[17];
    const float* Alog_bk  = (const float*)d_in[18];
    const float* D_bk     = (const float*)d_in[19];
    const float* xproj_bk = (const float*)d_in[20];
    const float* dtw_bk   = (const float*)d_in[21];
    const float* dtb_bk   = (const float*)d_in[22];
    const float* outproj_w= (const float*)d_in[23];
    const float* fng      = (const float*)d_in[24];
    const float* fnb      = (const float*)d_in[25];
    const float* fin_w    = (const float*)d_in[26];
    const float* fin_b    = (const float*)d_in[27];
    float* out = (float*)d_out;

    // ---------------- workspace layout ----------------
    float* ws   = (float*)d_ws;
    float* te   = ws;  ws += BB * DD;
    float* h    = ws;  ws += (size_t)NTOK * DD;
    float* xz   = ws;  ws += (size_t)NTOK * 2 * DI;
    float* u    = ws;  ws += (size_t)NTOK * DI;
    float* proj = ws;  ws += (size_t)NTOK * 64;
    float* dt   = ws;  ws += (size_t)NTOK * DI;
    float* y    = ws;  ws += (size_t)NTOK * DI;
    float* ytmp = ws;  ws += (size_t)NTOK * DD;
    float* o    = ws;  ws += (size_t)NTOK * 48;

    bf16* bws      = (bf16*)ws;
    bf16* hn_bf    = bws;  bws += (size_t)NTOK * DD;
    bf16* u_bf     = bws;  bws += (size_t)NTOK * DI;
    bf16* proj_bf  = bws;  bws += (size_t)NTOK * 64;
    bf16* y_bf     = bws;  bws += (size_t)NTOK * DI;
    bf16* inproj_bf= bws;  bws += (size_t)NLAY * 2 * DI * DD;
    bf16* xprojf_bf= bws;  bws += (size_t)NLAY * 64 * DI;
    bf16* xprojb_bf= bws;  bws += (size_t)NLAY * 64 * DI;
    bf16* dtwf_bf  = bws;  bws += (size_t)NLAY * DI * DR;
    bf16* dtwb_bf  = bws;  bws += (size_t)NLAY * DI * DR;
    bf16* outp_bf  = bws;  bws += (size_t)NLAY * DD * DI;
    bf16* fin_bf   = bws;  bws += (size_t)48 * DD;

    // ---------------- per-call weight conversion ----------------
    cvt(inproj_w,  inproj_bf, NLAY * 2 * DI * DD, stream);
    cvt(xproj_f,   xprojf_bf, NLAY * 64 * DI,     stream);
    cvt(xproj_bk,  xprojb_bf, NLAY * 64 * DI,     stream);
    cvt(dtw_f,     dtwf_bf,   NLAY * DI * DR,     stream);
    cvt(dtw_bk,    dtwb_bf,   NLAY * DI * DR,     stream);
    cvt(outproj_w, outp_bf,   NLAY * DD * DI,     stream);
    cvt(fin_w,     fin_bf,    48 * DD,            stream);

    patch_embed_kernel<<<NTOK, 256, 0, stream>>>(x, patch_w, patch_b, h);
    time_embed_kernel<<<BB, 256, 0, stream>>>(t, tw1, tb1, tw2, tb2, te);

    for (int i = 0; i < NLAY; ++i) {
        add_ln_kernel<<<NTOK, 256, 0, stream>>>(h, te, norm_g + i * DD,
                                                norm_b + i * DD, hn_bf);
        // xz = hn @ inproj_w[i].T         [4096, 1024]
        launch_gemm(hn_bf, DD, inproj_bf + (size_t)i * 2 * DI * DD, DD,
                    xz, nullptr, 2 * DI, NTOK, 2 * DI, DD, nullptr, 0, stream);
        // depthwise causal conv + silu    -> u (fp32 + bf16)
        conv_silu_kernel<<<dim3(DI / 256, NTOK), 256, 0, stream>>>(
            xz, conv_w + (size_t)i * DI * 4, conv_b + (size_t)i * DI, u, u_bf);

        // ---- forward scan ----
        launch_gemm(u_bf, DI, xprojf_bf + (size_t)i * 64 * DI, DI,
                    proj, proj_bf, 64, NTOK, 64, DI, nullptr, 0, stream);
        launch_gemm(proj_bf, 64, dtwf_bf + (size_t)i * DI * DR, DR,
                    dt, nullptr, DI, NTOK, DI, DR,
                    dtb_f + (size_t)i * DI, 1, stream);
        ssm_scan_kernel<<<dim3(DI / 256, BB), 256, 0, stream>>>(
            u, dt, proj, Alog_f + (size_t)i * DI * DS, D_f + (size_t)i * DI,
            y, 0, 0);

        // ---- backward scan (same token-order projections, reversed time) ----
        launch_gemm(u_bf, DI, xprojb_bf + (size_t)i * 64 * DI, DI,
                    proj, proj_bf, 64, NTOK, 64, DI, nullptr, 0, stream);
        launch_gemm(proj_bf, 64, dtwb_bf + (size_t)i * DI * DR, DR,
                    dt, nullptr, DI, NTOK, DI, DR,
                    dtb_bk + (size_t)i * DI, 1, stream);
        ssm_scan_kernel<<<dim3(DI / 256, BB), 256, 0, stream>>>(
            u, dt, proj, Alog_bk + (size_t)i * DI * DS, D_bk + (size_t)i * DI,
            y, 1, 1);

        // gate + output projection + residual
        gate_kernel<<<(NTOK * DI) / 256, 256, 0, stream>>>(y, xz, y_bf);
        launch_gemm(y_bf, DI, outp_bf + (size_t)i * DD * DI, DI,
                    ytmp, nullptr, DD, NTOK, DD, DI, nullptr, 0, stream);
        add_kernel<<<(NTOK * DD) / 256, 256, 0, stream>>>(h, ytmp, NTOK * DD);
    }

    // final LN + head + unpatchify
    add_ln_kernel<<<NTOK, 256, 0, stream>>>(h, nullptr, fng, fnb, hn_bf);
    launch_gemm(hn_bf, DD, fin_bf, DD, o, nullptr, 48, NTOK, 48, DD,
                fin_b, 0, stream);
    unpatch_kernel<<<(BB * CC * IMG * IMG + 255) / 256, 256, 0, stream>>>(o, out);
}
